// AttentionHead_41008347742774
// MI455X (gfx1250) — compile-verified, matrix operand order checked by reading, not run on hardware
//
#include <hip/hip_runtime.h>
#include <hip/hip_bf16.h>

// MI455X / gfx1250, wave32. bf16 WMMA pipeline:
//   proj: q,k,v = X@W+b  (bf16 out; V stored transposed [B,64,S])
//   attn: flash attention, 32 keys/iter, S^T = K x Q^T trick so softmax is
//         lane-local + 1 shuffle and P^T lands directly in B-operand layout.
#define S_LEN 2048
#define D_MODEL 1024
#define HEAD 64
#define NQT (S_LEN / 16)                      // 128 query tiles

typedef __attribute__((ext_vector_type(16))) __bf16 v16bf;
typedef __attribute__((ext_vector_type(8)))  float  v8f;
typedef __attribute__((ext_vector_type(4)))  unsigned int u32x4;

union FragBF {
    v16bf  v;
    __bf16 e[16];
    u32x4  q[2];
};

// ---------------------------------------------------------------------------
// Projection: Y = X @ W + b  (M=8192, K=1024, N=64)
// 128 threads = 4 waves; wave w owns 16x16 tile at (blockIdx.x*16, w*16).
// transposed==0: Y is bf16 [8192,64] row-major.  transposed==1: Y is [4,64,2048].
// ---------------------------------------------------------------------------
__global__ __launch_bounds__(128) void proj_qkv_kernel(
    const float* __restrict__ X, const float* __restrict__ W,
    const float* __restrict__ bias, __hip_bfloat16* __restrict__ Y,
    int transposed)
{
    __shared__ __bf16 Wlds[32 * 64];          // one K-slab of W, bf16

    const int tid  = threadIdx.x;
    const int lane = tid & 31;
    const int wv   = tid >> 5;
    const int m0   = blockIdx.x * 16;
    const int n0   = wv * 16;
    const int hl   = lane >> 4;               // half-wave select
    const int l15  = lane & 15;

    const float* Xrow = X + (size_t)(m0 + l15) * D_MODEL + hl * 8;

    v8f acc = {};

    for (int k0 = 0; k0 < D_MODEL; k0 += 32) {
        // stage W[k0:k0+32][0:64] into LDS as bf16 (coalesced float4 loads)
        const float* Wt = W + (size_t)k0 * HEAD;
#pragma unroll
        for (int i = 0; i < 4; ++i) {
            int idx4 = tid + i * 128;                     // 512 float4s total
            float4 w4 = ((const float4*)Wt)[idx4];
            int base = idx4 * 4;
            Wlds[base + 0] = (__bf16)w4.x;
            Wlds[base + 1] = (__bf16)w4.y;
            Wlds[base + 2] = (__bf16)w4.z;
            Wlds[base + 3] = (__bf16)w4.w;
        }
        __syncthreads();

        // A fragment (16x32 bf16): lane row = m0+l15, K split by half-wave
        FragBF a;
        float4 x0 = *(const float4*)(Xrow + k0);
        float4 x1 = *(const float4*)(Xrow + k0 + 4);
        float4 x2 = *(const float4*)(Xrow + k0 + 16);
        float4 x3 = *(const float4*)(Xrow + k0 + 20);
        a.e[0]=(__bf16)x0.x; a.e[1]=(__bf16)x0.y; a.e[2]=(__bf16)x0.z; a.e[3]=(__bf16)x0.w;
        a.e[4]=(__bf16)x1.x; a.e[5]=(__bf16)x1.y; a.e[6]=(__bf16)x1.z; a.e[7]=(__bf16)x1.w;
        a.e[8]=(__bf16)x2.x; a.e[9]=(__bf16)x2.y; a.e[10]=(__bf16)x2.z; a.e[11]=(__bf16)x2.w;
        a.e[12]=(__bf16)x3.x; a.e[13]=(__bf16)x3.y; a.e[14]=(__bf16)x3.z; a.e[15]=(__bf16)x3.w;

        // B fragment (32x16 bf16): lane col = n0+l15, K pattern mirrors A
        FragBF bfr;
        const int kb   = hl * 8;
        const int ncol = n0 + l15;
#pragma unroll
        for (int e = 0; e < 8; ++e) {
            bfr.e[e]     = Wlds[(kb + e) * HEAD + ncol];
            bfr.e[e + 8] = Wlds[(kb + 16 + e) * HEAD + ncol];
        }

        acc = __builtin_amdgcn_wmma_f32_16x16x32_bf16(
                  false, a.v, false, bfr.v, (short)0, acc, false, false);
        __syncthreads();
    }

    const float bn = bias[n0 + l15];
    if (!transposed) {
        // D layout: lane col n0+l15; rows m0 + hl*8 + r
        __hip_bfloat16* yp = Y + (size_t)(m0 + hl * 8) * HEAD + (n0 + l15);
#pragma unroll
        for (int r = 0; r < 8; ++r)
            yp[(size_t)r * HEAD] = __float2bfloat16(acc[r] + bn);
    } else {
        // V^T[b][h][s]: h = n0+l15, s = (m0+hl*8)+r consecutive -> one b128 store
        const int m = m0 + hl * 8;
        const int b = m >> 11;
        const int s = m & (S_LEN - 1);
        union { __hip_bfloat16 h[8]; u32x4 q; } pack;
#pragma unroll
        for (int r = 0; r < 8; ++r)
            pack.h[r] = __float2bfloat16(acc[r] + bn);
        *(u32x4*)(Y + ((size_t)b * HEAD + (n0 + l15)) * S_LEN + s) = pack.q;
    }
}

// ---------------------------------------------------------------------------
// Flash attention, causal, 32 keys per iteration. One wave per 16-query tile.
//   scores:  S^T(2x 16x16) = K_tile x Q^T           (4 bf16 WMMAs)
//   update:  O^T(16x16 per h-chunk) += V^T x P^T    (4 FULL-k bf16 WMMAs)
// Softmax in exp2 domain; per-query stats are lane-local + one shfl_xor(16).
// ---------------------------------------------------------------------------
__global__ __launch_bounds__(256) void attn_kernel(
    const __hip_bfloat16* __restrict__ Qb,   // [B,S,64] bf16
    const __hip_bfloat16* __restrict__ Kb,   // [B,S,64] bf16
    const __hip_bfloat16* __restrict__ Vt,   // [B,64,S] bf16
    float* __restrict__ Out)                 // [B,S,64] f32
{
    const int lane = threadIdx.x & 31;
    const int wv   = threadIdx.x >> 5;
    // work-balancing swizzle: pair short (early) and long (late) query tiles
    const int idx   = blockIdx.x * 8 + wv;                       // 0..127
    const int qtile = (idx & 1) ? (NQT - 1 - (idx >> 1)) : (idx >> 1);
    const int b     = blockIdx.y;
    const int hl    = lane >> 4;
    const int l15   = lane & 15;
    const int q0    = qtile * 16;

    const __hip_bfloat16* qp = Qb + (size_t)b * S_LEN * HEAD;
    const __hip_bfloat16* kp = Kb + (size_t)b * S_LEN * HEAD;
    const __hip_bfloat16* vp = Vt + (size_t)b * HEAD * S_LEN;
    float*                op = Out + (size_t)b * S_LEN * HEAD;

    // Q^T B-fragments, chunk c covers h = c*32..c*32+31
    FragBF qt0, qt1;
    {
        const u32x4* qrow = (const u32x4*)(qp + (size_t)(q0 + l15) * HEAD);
        qt0.q[0] = qrow[hl];     qt0.q[1] = qrow[hl + 2];
        qt1.q[0] = qrow[4 + hl]; qt1.q[1] = qrow[6 + hl];
    }

    v8f o[4] = {{}, {}, {}, {}};
    float m_run = -1e30f, l_run = 0.0f;
    const int qg = q0 + l15;                  // this lane's query index
    const float SC = 0.125f * 1.44269504088896340736f;  // 1/sqrt(64) * log2(e)

    const int jmax = qtile >> 1;              // 32-key blocks
    for (int j = 0; j <= jmax; ++j) {
        const int base = j * 32;
        if (j < jmax) {
            __builtin_prefetch(kp + (size_t)(base + 32) * HEAD, 0, 0);
            __builtin_prefetch(vp + (size_t)base + 32, 0, 0);
        }

        // K A-fragments for the two 16-key tiles
        const u32x4* kr0 = (const u32x4*)(kp + (size_t)(base + l15) * HEAD);
        const u32x4* kr1 = (const u32x4*)(kp + (size_t)(base + 16 + l15) * HEAD);
        FragBF k0a, k0b, k1a, k1b;
        k0a.q[0] = kr0[hl];     k0a.q[1] = kr0[hl + 2];
        k0b.q[0] = kr0[4 + hl]; k0b.q[1] = kr0[6 + hl];
        k1a.q[0] = kr1[hl];     k1a.q[1] = kr1[hl + 2];
        k1b.q[0] = kr1[4 + hl]; k1b.q[1] = kr1[6 + hl];

        v8f s0 = {}, s1 = {};
        s0 = __builtin_amdgcn_wmma_f32_16x16x32_bf16(
                 false, k0a.v, false, qt0.v, (short)0, s0, false, false);
        s0 = __builtin_amdgcn_wmma_f32_16x16x32_bf16(
                 false, k0b.v, false, qt1.v, (short)0, s0, false, false);
        s1 = __builtin_amdgcn_wmma_f32_16x16x32_bf16(
                 false, k1a.v, false, qt0.v, (short)0, s1, false, false);
        s1 = __builtin_amdgcn_wmma_f32_16x16x32_bf16(
                 false, k1b.v, false, qt1.v, (short)0, s1, false, false);

        // scale into exp2 domain, causal mask, tile max
        float p[16];
        float mt = -1e30f;
#pragma unroll
        for (int r = 0; r < 8; ++r) {
            float t0 = s0[r] * SC;
            float t1 = s1[r] * SC;
            const int kg0 = base + hl * 8 + r;
            const int kg1 = kg0 + 16;
            t0 = (kg0 <= qg) ? t0 : -1e30f;
            t1 = (kg1 <= qg) ? t1 : -1e30f;
            p[r]     = t0;
            p[r + 8] = t1;
            mt = fmaxf(mt, fmaxf(t0, t1));
        }
        mt = fmaxf(mt, __shfl_xor(mt, 16, 32));

        const float m_new = fmaxf(m_run, mt);
        const float corr  = exp2f(m_run - m_new);
        m_run = m_new;

        FragBF pb;                            // P^T as full 32-k B operand
        float lt = 0.0f;
#pragma unroll
        for (int e = 0; e < 16; ++e) {
            const float pe = exp2f(p[e] - m_new);
            lt += pe;
            pb.e[e] = (__bf16)pe;
        }
        lt += __shfl_xor(lt, 16, 32);
        l_run = l_run * corr + lt;

#pragma unroll
        for (int c = 0; c < 4; ++c)
#pragma unroll
            for (int r = 0; r < 8; ++r)
                o[c][r] *= corr;

        // O^T += V^T x P^T, 4 h-chunks of 16, full k=32
        const size_t vc0 = (size_t)(base + hl * 8);
#pragma unroll
        for (int c = 0; c < 4; ++c) {
            const __hip_bfloat16* vrow = vp + (size_t)(c * 16 + l15) * S_LEN;
            FragBF va;
            va.q[0] = *(const u32x4*)(vrow + vc0);
            va.q[1] = *(const u32x4*)(vrow + vc0 + 16);
            o[c] = __builtin_amdgcn_wmma_f32_16x16x32_bf16(
                       false, va.v, false, pb.v, (short)0, o[c], false, false);
        }
    }

    // normalize + store: lane col q = q0+l15; chunk c rows h = c*16 + hl*8 + r
    const float inv = 1.0f / l_run;
    float* dst0 = op + (size_t)(q0 + l15) * HEAD + hl * 8;
#pragma unroll
    for (int c = 0; c < 4; ++c) {
        float* dst = dst0 + c * 16;
        *(float4*)dst =
            make_float4(o[c][0] * inv, o[c][1] * inv, o[c][2] * inv, o[c][3] * inv);
        *(float4*)(dst + 4) =
            make_float4(o[c][4] * inv, o[c][5] * inv, o[c][6] * inv, o[c][7] * inv);
    }
}

// ---------------------------------------------------------------------------
extern "C" void kernel_launch(void* const* d_in, const int* in_sizes, int n_in,
                              void* d_out, int out_size, void* d_ws, size_t ws_size,
                              hipStream_t stream) {
    const float* query = (const float*)d_in[0];
    const float* key   = (const float*)d_in[1];
    const float* value = (const float*)d_in[2];
    // d_in[3] = mask: causal, handled analytically
    const float* Wq = (const float*)d_in[4];
    const float* bq = (const float*)d_in[5];
    const float* Wk = (const float*)d_in[6];
    const float* bk = (const float*)d_in[7];
    const float* Wv = (const float*)d_in[8];
    const float* bv = (const float*)d_in[9];
    float* out = (float*)d_out;

    char* ws = (char*)d_ws;                  // 3 MB used
    __hip_bfloat16* qbf = (__hip_bfloat16*)(ws);
    __hip_bfloat16* kbf = (__hip_bfloat16*)(ws + (1u << 20));
    __hip_bfloat16* vtb = (__hip_bfloat16*)(ws + (2u << 20));

    const int Mtiles = (4 * S_LEN) / 16;     // 512
    proj_qkv_kernel<<<Mtiles, 128, 0, stream>>>(query, Wq, bq, qbf, 0);
    proj_qkv_kernel<<<Mtiles, 128, 0, stream>>>(key,   Wk, bk, kbf, 0);
    proj_qkv_kernel<<<Mtiles, 128, 0, stream>>>(value, Wv, bv, vtb, 1);

    attn_kernel<<<dim3(NQT / 8, 4), 256, 0, stream>>>(qbf, kbf, vtb, out);
}
